// TransformerBlock_70042326663714
// MI455X (gfx1250) — matrix operand order repaired and checked by reading, not compiled, not run on hardware
//
#include <hip/hip_runtime.h>
#include <hip/hip_bf16.h>
#include <math.h>

#define EMBED 1024
#define FFDIM 4096
#define HEADS 16
#define HDIM  64
#define SEQ   2048
#define BATCH 2
#define ROWS  (BATCH * SEQ)   // 4096 token rows

typedef __attribute__((ext_vector_type(16))) __bf16 v16bf;
typedef __attribute__((ext_vector_type(8)))  float  v8f;
typedef __attribute__((ext_vector_type(4)))  int    v4i_;

// ---------------------------------------------------------------- helpers

__device__ __forceinline__ unsigned short f2bf(float x) {
  union { float f; unsigned u; } c; c.f = x;
  unsigned r = c.u + 0x7FFFu + ((c.u >> 16) & 1u);   // round-to-nearest-even
  return (unsigned short)(r >> 16);
}

union FragB { v16bf v; unsigned u[8]; };

// Async 16-byte global->LDS copy (GLOBAL_LOAD_ASYNC_TO_LDS_B128, ASYNCcnt).
__device__ __forceinline__ void async_b128(void* lds, const void* g) {
#if defined(__gfx1250__) && __has_builtin(__builtin_amdgcn_global_load_async_to_lds_b128)
  typedef __attribute__((address_space(1))) v4i_ gq_t;   // global int4
  typedef __attribute__((address_space(3))) v4i_ lq_t;   // LDS int4
  __builtin_amdgcn_global_load_async_to_lds_b128((gq_t*)g, (lq_t*)lds, 0, 0);
#else
  *(uint4*)lds = *(const uint4*)g;
#endif
}

__device__ __forceinline__ void wait_async0() {
#if defined(__gfx1250__) && __has_builtin(__builtin_amdgcn_s_wait_asynccnt)
  __builtin_amdgcn_s_wait_asynccnt(0);
#elif defined(__gfx1250__)
  asm volatile("s_wait_asynccnt 0" ::: "memory");
#endif
}

// A-matrix fragment (16x32 bf16, MxK) from a row-major [m][k] LDS tile.
// ISA layout: lanes 0-15: M=lane, VGPR0-3 K=0..7 pairs, VGPR4-7 K=16..23;
//             lanes16-31: M=lane-16, K=8..15 and K=24..31.
__device__ __forceinline__ v16bf lds_frag_A(const unsigned short* base,
                                            int m0, int ldk, int k0) {
  const int lane = threadIdx.x & 31;
  const int half = lane >> 4;
  const int m    = (lane & 15) + m0;
  const unsigned* row = (const unsigned*)(base + m * ldk);
  FragB r;
#pragma unroll
  for (int j = 0; j < 4; ++j) r.u[j]     = row[(k0 + half * 8) / 2 + j];
#pragma unroll
  for (int j = 0; j < 4; ++j) r.u[4 + j] = row[(k0 + 16 + half * 8) / 2 + j];
  return r.v;
}

// B-matrix fragment (32x16 bf16, KxN) from a row-major [n][k] LDS tile (B^T).
// ISA layout: lanes 0-15: N=lane, K=0..7 and K=16..23; lanes 16-31: N=lane-16, K=8..15 and K=24..31.
__device__ __forceinline__ v16bf lds_frag_B(const unsigned short* base,
                                            int n0, int ldk, int k0) {
  const int lane = threadIdx.x & 31;
  const int half = lane >> 4;
  const int n    = (lane & 15) + n0;
  const unsigned* row = (const unsigned*)(base + n * ldk);
  FragB r;
#pragma unroll
  for (int j = 0; j < 4; ++j) r.u[j]     = row[(k0 + half * 8) / 2 + j];
#pragma unroll
  for (int j = 0; j < 4; ++j) r.u[4 + j] = row[(k0 + 16 + half * 8) / 2 + j];
  return r.v;
}

__device__ __forceinline__ v8f wmma_bf16(v16bf a, v16bf b, v8f c) {
  return __builtin_amdgcn_wmma_f32_16x16x32_bf16(
      /*neg_a=*/false, a, /*neg_b=*/false, b,
      /*c_mod=*/(short)0, c, /*reuse_a=*/false, /*reuse_b=*/false);
}

// ------------------------------------------- fp32 [K][N] -> bf16 [N][K] (W^T)

__global__ __launch_bounds__(256) void cvt_transpose_bf16(
    const float* __restrict__ in, unsigned short* __restrict__ out,
    int K, int N) {
  __shared__ unsigned short tile[32][33];
  const int k0 = blockIdx.y * 32;
  const int n0 = blockIdx.x * 32;
  const int tx = threadIdx.x;          // 0..31
  const int ty = threadIdx.y;          // 0..7
#pragma unroll
  for (int i = 0; i < 32; i += 8)
    tile[ty + i][tx] = f2bf(in[(size_t)(k0 + ty + i) * N + n0 + tx]);
  __syncthreads();
#pragma unroll
  for (int i = 0; i < 32; i += 8)
    out[(size_t)(n0 + ty + i) * K + k0 + tx] = tile[tx][ty + i];
}

// ---------------------------------------------------------------- layernorm -> bf16

__global__ __launch_bounds__(256) void ln_kernel(const float* __restrict__ x,
                                                 const float* __restrict__ g,
                                                 const float* __restrict__ b,
                                                 unsigned short* __restrict__ out) {
  __shared__ float s1[256];
  __shared__ float s2[256];
  const int row = blockIdx.x;
  const int t = threadIdx.x;
  const float4 xv = *(const float4*)(x + (size_t)row * EMBED + t * 4);
  float sum = xv.x + xv.y + xv.z + xv.w;
  float sq  = xv.x * xv.x + xv.y * xv.y + xv.z * xv.z + xv.w * xv.w;
  s1[t] = sum; s2[t] = sq;
  __syncthreads();
  for (int off = 128; off > 0; off >>= 1) {
    if (t < off) { s1[t] += s1[t + off]; s2[t] += s2[t + off]; }
    __syncthreads();
  }
  const float mu  = s1[0] * (1.0f / EMBED);
  const float var = s2[0] * (1.0f / EMBED) - mu * mu;
  const float rs  = rsqrtf(var + 1e-5f);
  const int c = t * 4;
  const float4 gv = *(const float4*)(g + c);
  const float4 bv = *(const float4*)(b + c);
  unsigned short* o = out + (size_t)row * EMBED + c;
  o[0] = f2bf((xv.x - mu) * rs * gv.x + bv.x);
  o[1] = f2bf((xv.y - mu) * rs * gv.y + bv.y);
  o[2] = f2bf((xv.z - mu) * rs * gv.z + bv.z);
  o[3] = f2bf((xv.w - mu) * rs * gv.w + bv.w);
}

// ---------------------------------------------------------------- WMMA GEMM
// C[M,N] = A[M,K](bf16) @ W(bf16, stored transposed [N][K]) + bias.
// 256 threads = 8 waves; block tile 128x128, K-step 32 (one WMMA).
// Both tiles staged with GLOBAL_LOAD_ASYNC_TO_LDS_B128.
// TRV=1: scatter output as per-head-transposed V ([b][h][d][s]) for attention.

template <int GELU, int RES, int OBF, int TRV>
__global__ __launch_bounds__(256) void gemm_bf16(
    const unsigned short* __restrict__ A, const unsigned short* __restrict__ WT,
    const float* __restrict__ bias, const float* __restrict__ resid,
    unsigned short* __restrict__ outB, float* __restrict__ outF,
    int M, int N, int K) {
  __shared__ __align__(16) unsigned short As[128 * 32];   // [m][k]
  __shared__ __align__(16) unsigned short Bs[128 * 32];   // [n][k]
  const int t = threadIdx.x;
  const int wid = t >> 5;
  const int wm = wid >> 1;        // 0..3
  const int wn = wid & 1;         // 0..1
  const int row0 = blockIdx.y * 128;
  const int col0 = blockIdx.x * 128;

  v8f acc[2][4];
#pragma unroll
  for (int i = 0; i < 2; ++i)
#pragma unroll
    for (int j = 0; j < 4; ++j)
#pragma unroll
      for (int e = 0; e < 8; ++e) acc[i][j][e] = 0.0f;

  const int nK = K >> 5;
  for (int kt = 0; kt < nK; ++kt) {
    // async-stage A tile [128][32] and B^T tile [128][32] (both k-contiguous)
#pragma unroll
    for (int p = 0; p < 2; ++p) {
      const int i0 = (p * 256 + t) * 8;
      const int r = i0 >> 5, c = i0 & 31;
      async_b128(As + r * 32 + c, A  + (size_t)(row0 + r) * K + kt * 32 + c);
      async_b128(Bs + r * 32 + c, WT + (size_t)(col0 + r) * K + kt * 32 + c);
    }
    if (kt + 1 < nK)
      __builtin_prefetch(A + (size_t)(row0 + (t >> 1)) * K + (kt + 1) * 32, 0, 0);
    wait_async0();
    __syncthreads();

    const v16bf a0 = lds_frag_A(As, wm * 32, 32, 0);
    const v16bf a1 = lds_frag_A(As, wm * 32 + 16, 32, 0);
#pragma unroll
    for (int nt = 0; nt < 4; ++nt) {
      const v16bf bfr = lds_frag_B(Bs, wn * 64 + nt * 16, 32, 0);
      acc[0][nt] = wmma_bf16(a0, bfr, acc[0][nt]);
      acc[1][nt] = wmma_bf16(a1, bfr, acc[1][nt]);
    }
    __syncthreads();
  }

  // epilogue: C layout — lane half selects M-group, VGPR j -> m = j + 8*half
  const int lane = t & 31;
  const int half = lane >> 4;
  const int ncol = lane & 15;
#pragma unroll
  for (int im = 0; im < 2; ++im)
#pragma unroll
    for (int nt = 0; nt < 4; ++nt)
#pragma unroll
      for (int j = 0; j < 8; ++j) {
        const int r = row0 + wm * 32 + im * 16 + half * 8 + j;
        const int c = col0 + wn * 64 + nt * 16 + ncol;
        float v = acc[im][nt][j] + bias[c];
        if (GELU) {
          const float x3 = v * v * v;
          v = 0.5f * v * (1.0f + tanhf(0.7978845608028654f * (v + 0.044715f * x3)));
        }
        if (TRV) {
          // scatter as V^T per head: [b][h][d][s]
          const int bb = r >> 11, s = r & 2047;
          const int hh = c >> 6,  dd = c & 63;
          outB[((size_t)((bb * HEADS + hh) * HDIM + dd)) * SEQ + s] = f2bf(v);
        } else {
          const size_t idx = (size_t)r * N + c;
          float vv = v;
          if (RES) vv += resid[idx];
          if (OBF) outB[idx] = f2bf(vv);
          else     outF[idx] = vv;
        }
      }
}

// ---------------------------------------------------------------- flash attention
// One block (128 thr = 4 waves) per (batch, head, 64-row q tile).
// Q/K/V tiles async-staged to LDS; scores WMMA -> LDS, online softmax, P@V WMMA.
// V arrives pre-transposed per head ([b][h][d][s]) so its tile is k-contiguous.

__global__ __launch_bounds__(128) void attn_kernel(
    const unsigned short* __restrict__ Q, const unsigned short* __restrict__ Kb,
    const unsigned short* __restrict__ VT, unsigned short* __restrict__ O) {
  __shared__ __align__(16) unsigned short Qs[64 * 64];  // [m][d]
  __shared__ __align__(16) unsigned short Ks[64 * 64];  // [key][d]  (B^T for scores)
  __shared__ __align__(16) unsigned short Vt[64 * 64];  // [d][key]  (B^T for P@V)
  __shared__ __align__(16) unsigned short Ps[64 * 64];  // [m][key]
  __shared__ float Ss[64 * 64];                         // [m][key] scores
  __shared__ float mrow[64], lrow[64], crow[64];

  const int t = threadIdx.x;
  const int wid = t >> 5;
  const int lane = t & 31;
  const int half = lane >> 4;
  const int ncol = lane & 15;
  const int bid = blockIdx.x;
  const int qb = bid & 31;            // SEQ/64 = 32 q tiles
  const int h  = (bid >> 5) & 15;
  const int b  = bid >> 9;
  const int q0 = qb * 64;
  const size_t headoff = (size_t)h * HDIM;
  const size_t vhead   = (size_t)(b * HEADS + h) * HDIM;   // row base in VT

  // async-stage Q tile (once)
#pragma unroll
  for (int p = 0; p < 4; ++p) {
    const int i0 = (p * 128 + t) * 8;
    const int r = i0 >> 6, c = i0 & 63;
    async_b128(Qs + r * 64 + c,
               Q + (size_t)(b * SEQ + q0 + r) * EMBED + headoff + c);
  }
  if (t < 64) { mrow[t] = -1e30f; lrow[t] = 0.0f; }

  const int m0 = wid * 16;
  v8f acc[4];
#pragma unroll
  for (int nt = 0; nt < 4; ++nt)
#pragma unroll
    for (int e = 0; e < 8; ++e) acc[nt][e] = 0.0f;

  for (int kt = 0; kt < SEQ / 64; ++kt) {
    const int k0g = kt * 64;
    // K tile [key][d] and V^T tile [d][key] — both contiguous 16B chunks
#pragma unroll
    for (int p = 0; p < 4; ++p) {
      const int i0 = (p * 128 + t) * 8;
      const int r = i0 >> 6, c = i0 & 63;
      async_b128(Ks + r * 64 + c,
                 Kb + (size_t)(b * SEQ + k0g + r) * EMBED + headoff + c);
      async_b128(Vt + r * 64 + c,
                 VT + (vhead + r) * SEQ + k0g + c);
    }
    wait_async0();
    __syncthreads();

    // scores S = Q @ K^T (per wave: 16 rows x 64 keys)
    v8f sc[4];
#pragma unroll
    for (int nt = 0; nt < 4; ++nt)
#pragma unroll
      for (int e = 0; e < 8; ++e) sc[nt][e] = 0.0f;
#pragma unroll
    for (int kd = 0; kd < HDIM; kd += 32) {
      const v16bf aq = lds_frag_A(Qs, m0, 64, kd);
#pragma unroll
      for (int nt = 0; nt < 4; ++nt) {
        const v16bf bk = lds_frag_B(Ks, nt * 16, 64, kd);
        sc[nt] = wmma_bf16(aq, bk, sc[nt]);
      }
    }
#pragma unroll
    for (int nt = 0; nt < 4; ++nt)
#pragma unroll
      for (int j = 0; j < 8; ++j)
        Ss[(m0 + half * 8 + j) * 64 + nt * 16 + ncol] = sc[nt][j] * 0.125f;
    __syncthreads();

    // online softmax (one thread per q row)
    if (t < 64) {
      const float mo = mrow[t];
      float mn = mo;
      const float* sr = Ss + t * 64;
#pragma unroll 8
      for (int j = 0; j < 64; ++j) mn = fmaxf(mn, sr[j]);
      const float cf = __expf(mo - mn);
      float l = lrow[t] * cf;
      unsigned short* pr = Ps + t * 64;
      for (int j = 0; j < 64; ++j) {
        const float p = __expf(sr[j] - mn);
        l += p;
        pr[j] = f2bf(p);
      }
      mrow[t] = mn; lrow[t] = l; crow[t] = cf;
    }
    __syncthreads();

    // rescale accumulator, then acc += P @ V
#pragma unroll
    for (int nt = 0; nt < 4; ++nt)
#pragma unroll
      for (int j = 0; j < 8; ++j) acc[nt][j] *= crow[m0 + half * 8 + j];
#pragma unroll
    for (int kk = 0; kk < 64; kk += 32) {
      const v16bf ap = lds_frag_A(Ps, m0, 64, kk);
#pragma unroll
      for (int nt = 0; nt < 4; ++nt) {
        const v16bf bv = lds_frag_B(Vt, nt * 16, 64, kk);
        acc[nt] = wmma_bf16(ap, bv, acc[nt]);
      }
    }
    __syncthreads();
  }

  // write normalized output (bf16)
#pragma unroll
  for (int nt = 0; nt < 4; ++nt)
#pragma unroll
    for (int j = 0; j < 8; ++j) {
      const int m = m0 + half * 8 + j;
      const float inv = 1.0f / lrow[m];
      const int c = nt * 16 + ncol;
      O[(size_t)(b * SEQ + q0 + m) * EMBED + headoff + c] = f2bf(acc[nt][j] * inv);
    }
}

// ---------------------------------------------------------------- launcher

extern "C" void kernel_launch(void* const* d_in, const int* in_sizes, int n_in,
                              void* d_out, int out_size, void* d_ws, size_t ws_size,
                              hipStream_t stream) {
  const float* x     = (const float*)d_in[0];
  const float* Wq    = (const float*)d_in[1];
  const float* bq    = (const float*)d_in[2];
  const float* Wk    = (const float*)d_in[3];
  const float* bk    = (const float*)d_in[4];
  const float* Wv    = (const float*)d_in[5];
  const float* bv    = (const float*)d_in[6];
  const float* Wo    = (const float*)d_in[7];
  const float* bo    = (const float*)d_in[8];
  const float* W1    = (const float*)d_in[9];
  const float* b1    = (const float*)d_in[10];
  const float* W2    = (const float*)d_in[11];
  const float* b2    = (const float*)d_in[12];
  const float* ln1g  = (const float*)d_in[13];
  const float* ln1b  = (const float*)d_in[14];
  const float* ln2g  = (const float*)d_in[15];
  const float* ln2b  = (const float*)d_in[16];
  float* out = (float*)d_out;

  char* ws = (char*)d_ws;
  size_t off = 0;
  auto salloc = [&](size_t bytes) -> void* {
    void* p = ws + off;
    off += (bytes + 255) & ~(size_t)255;
    return p;
  };
  const size_t EE = (size_t)EMBED * EMBED;     // 1M
  const size_t EF = (size_t)EMBED * FFDIM;     // 4M
  const size_t RE = (size_t)ROWS * EMBED;      // 4M
  const size_t RF = (size_t)ROWS * FFDIM;      // 16M

  unsigned short* wq_t   = (unsigned short*)salloc(EE * 2);   // [N][K] bf16
  unsigned short* wk_t   = (unsigned short*)salloc(EE * 2);
  unsigned short* wv_t   = (unsigned short*)salloc(EE * 2);
  unsigned short* wo_t   = (unsigned short*)salloc(EE * 2);
  unsigned short* w1_t   = (unsigned short*)salloc(EF * 2);
  unsigned short* w2_t   = (unsigned short*)salloc(EF * 2);
  unsigned short* ln1_bf = (unsigned short*)salloc(RE * 2);
  unsigned short* q_bf   = (unsigned short*)salloc(RE * 2);
  unsigned short* k_bf   = (unsigned short*)salloc(RE * 2);
  unsigned short* vT_bf  = (unsigned short*)salloc(RE * 2);   // [b][h][d][s]
  unsigned short* at_bf  = (unsigned short*)salloc(RE * 2);
  float*          x1     = (float*)salloc(RE * 4);
  unsigned short* ln2_bf = (unsigned short*)salloc(RE * 2);
  unsigned short* h_bf   = (unsigned short*)salloc(RF * 2);

  // weight convert + transpose (deterministic every call): W[K][N] -> Wt[N][K]
  dim3 tb(32, 8);
  cvt_transpose_bf16<<<dim3(EMBED / 32, EMBED / 32), tb, 0, stream>>>(Wq, wq_t, EMBED, EMBED);
  cvt_transpose_bf16<<<dim3(EMBED / 32, EMBED / 32), tb, 0, stream>>>(Wk, wk_t, EMBED, EMBED);
  cvt_transpose_bf16<<<dim3(EMBED / 32, EMBED / 32), tb, 0, stream>>>(Wv, wv_t, EMBED, EMBED);
  cvt_transpose_bf16<<<dim3(EMBED / 32, EMBED / 32), tb, 0, stream>>>(Wo, wo_t, EMBED, EMBED);
  cvt_transpose_bf16<<<dim3(FFDIM / 32, EMBED / 32), tb, 0, stream>>>(W1, w1_t, EMBED, FFDIM);
  cvt_transpose_bf16<<<dim3(EMBED / 32, FFDIM / 32), tb, 0, stream>>>(W2, w2_t, FFDIM, EMBED);

  // LN1
  ln_kernel<<<ROWS, 256, 0, stream>>>(x, ln1g, ln1b, ln1_bf);

  // Q/K/V projections (V written per-head transposed)
  dim3 gE(EMBED / 128, ROWS / 128);     // (8, 32)
  gemm_bf16<0, 0, 1, 0><<<gE, 256, 0, stream>>>(ln1_bf, wq_t, bq, nullptr, q_bf,  nullptr, ROWS, EMBED, EMBED);
  gemm_bf16<0, 0, 1, 0><<<gE, 256, 0, stream>>>(ln1_bf, wk_t, bk, nullptr, k_bf,  nullptr, ROWS, EMBED, EMBED);
  gemm_bf16<0, 0, 1, 1><<<gE, 256, 0, stream>>>(ln1_bf, wv_t, bv, nullptr, vT_bf, nullptr, ROWS, EMBED, EMBED);

  // flash attention: BATCH * HEADS * (SEQ/64) = 1024 blocks of 128 threads
  attn_kernel<<<BATCH * HEADS * (SEQ / 64), 128, 0, stream>>>(q_bf, k_bf, vT_bf, at_bf);

  // out projection + residual -> x1 (fp32)
  gemm_bf16<0, 1, 0, 0><<<gE, 256, 0, stream>>>(at_bf, wo_t, bo, x, nullptr, x1, ROWS, EMBED, EMBED);

  // LN2
  ln_kernel<<<ROWS, 256, 0, stream>>>(x1, ln2g, ln2b, ln2_bf);

  // FFN1 + GELU -> h (bf16)
  dim3 gF(FFDIM / 128, ROWS / 128);     // (32, 32)
  gemm_bf16<1, 0, 1, 0><<<gF, 256, 0, stream>>>(ln2_bf, w1_t, b1, nullptr, h_bf, nullptr, ROWS, FFDIM, EMBED);

  // FFN2 + residual -> out (fp32)
  gemm_bf16<0, 1, 0, 0><<<gE, 256, 0, stream>>>(h_bf, w2_t, b2, x1, nullptr, out, ROWS, EMBED, FFDIM);

  (void)in_sizes; (void)n_in; (void)out_size; (void)ws_size;
}
